// NeuroVPR_Vanilla_SNN_86474871538275
// MI455X (gfx1250) — matrix operand order repaired and checked by reading, not compiled, not run on hardware
//
#include <hip/hip_runtime.h>
#include <hip/hip_fp16.h>

typedef _Float16 v16h __attribute__((ext_vector_type(16)));
typedef _Float16 v8h  __attribute__((ext_vector_type(8)));
typedef _Float16 v4h  __attribute__((ext_vector_type(4)));
typedef float    v8f  __attribute__((ext_vector_type(8)));
typedef float    v4f  __attribute__((ext_vector_type(4)));

#define BM   64
#define BK   64
#define LDSK 72   // padded f16 row stride for W tile in LDS (bank spread)

// One block: BM=64 batch rows x (NT*16) output neurons, K looped in BK=64 chunks.
// 8 waves, arranged 4 (M) x 2 (N): each wave owns 16 rows x (NT/2)*16 cols.
// A operand: loaded straight from global per the CDNA5 16-bit A-matrix layout
// (lane<16: K koff+0..7 & koff+16..23 with koff=(lane>>4)*8), f32->f16 on the fly.
// B operand: W rows ([N,K] row-major) staged via LDS; per-lane layout:
// same interleave as A: K ksub+kbB+0..7 & ksub+kbB+16..23 with kbB=(lane>>4)*8.
template<int NT, bool AF32>
__global__ __launch_bounds__(256, 1)
void gemm_lif(const void* __restrict__ Aptr, int lda,
              const float* __restrict__ W, int ldw, int wrows,
              const float* __restrict__ bias,
              float* __restrict__ vstate, int ldv,
              _Float16* __restrict__ sout, int ldsout,
              float* __restrict__ fout, int ldfout,
              int K, int Nreal)
{
    constexpr int TCOLS = NT / 2;
    __shared__ _Float16 ldsW[NT * 16 * LDSK];

    const int tid    = threadIdx.x;
    const int lane   = tid & 31;
    const int wm     = (tid >> 5) & 3;   // wave row group (0..3)
    const int wn     = tid >> 7;         // wave col group (0..1)
    const int blockM = blockIdx.x * BM;
    const int KT     = K / BK;

    const int nlane  = lane & 15;
    const int koffA  = (lane >> 4) * 8;   // A-operand K interleave
    const int kbB    = (lane >> 4) * 8;   // B-operand K interleave
    const int myrow  = blockM + wm * 16 + nlane;

    const float*    arowF = (const float*)Aptr    + (size_t)myrow * lda;
    const _Float16* arowH = (const _Float16*)Aptr + (size_t)myrow * lda;

    v8f acc[TCOLS] = {};
    v4f wreg[NT];
    const v4f vzero = {};

    auto loadW = [&](int kt) {
        const int k0 = kt * BK;
        #pragma unroll
        for (int i = 0; i < NT; ++i) {
            int flat = tid + i * 256;
            int n    = flat >> 4;       // W row (output neuron)
            int kv   = flat & 15;       // float4 index within BK
            if (n < wrows)
                wreg[i] = *(const v4f*)(W + (size_t)n * ldw + k0 + kv * 4);
            else
                wreg[i] = vzero;        // zero-pad (layer 3, N=100 -> 128)
        }
    };
    auto storeW = [&]() {
        #pragma unroll
        for (int i = 0; i < NT; ++i) {
            int flat = tid + i * 256;
            int n    = flat >> 4;
            int kv   = flat & 15;
            v4h h;
            #pragma unroll
            for (int j = 0; j < 4; ++j) h[j] = (_Float16)wreg[i][j];
            *(v4h*)(&ldsW[n * LDSK + kv * 4]) = h;
        }
    };
    auto compute = [&](int kt) {
        const int k0 = kt * BK;
        #pragma unroll
        for (int ks = 0; ks < 2; ++ks) {
            const int ksub = ks * 32;
            v16h a;
            if (AF32) {
                const float* p = arowF + k0 + ksub + koffA;
                v4f f0 = *(const v4f*)(p);
                v4f f1 = *(const v4f*)(p + 4);
                v4f f2 = *(const v4f*)(p + 16);
                v4f f3 = *(const v4f*)(p + 20);
                #pragma unroll
                for (int j = 0; j < 4; ++j) {
                    a[j]      = (_Float16)f0[j];
                    a[4 + j]  = (_Float16)f1[j];
                    a[8 + j]  = (_Float16)f2[j];
                    a[12 + j] = (_Float16)f3[j];
                }
            } else {
                const _Float16* p = arowH + k0 + ksub + koffA;
                v8h h0 = *(const v8h*)(p);
                v8h h1 = *(const v8h*)(p + 16);
                a = __builtin_shufflevector(h0, h1, 0,1,2,3,4,5,6,7,8,9,10,11,12,13,14,15);
            }
            #pragma unroll
            for (int ni = 0; ni < TCOLS; ++ni) {
                int nloc = wn * TCOLS * 16 + ni * 16 + nlane;
                const _Float16* bp = &ldsW[nloc * LDSK + ksub + kbB];
                v8h b0 = *(const v8h*)(bp);
                v8h b1 = *(const v8h*)(bp + 16);
                v16h b = __builtin_shufflevector(b0, b1, 0,1,2,3,4,5,6,7,8,9,10,11,12,13,14,15);
                acc[ni] = __builtin_amdgcn_wmma_f32_16x16x32_f16(
                    false, a, false, b, (short)0, acc[ni], false, false);
            }
        }
    };

    // software pipeline: global->regs prefetch overlaps WMMA on current LDS tile
    loadW(0);
    storeW();
    __syncthreads();
    for (int kt = 0; kt < KT; ++kt) {
        if (kt + 1 < KT) loadW(kt + 1);
        compute(kt);
        if (kt + 1 < KT) {
            __syncthreads();
            storeW();
            __syncthreads();
        }
    }

    // epilogue: bias + LIF (tau=2, v_th=1, hard reset to 0)
    const int rbase = blockM + wm * 16 + (lane >> 4) * 8;
    #pragma unroll
    for (int ni = 0; ni < TCOLS; ++ni) {
        int gcol = wn * TCOLS * 16 + ni * 16 + nlane;
        if (gcol < Nreal) {
            float bv = bias[gcol];
            #pragma unroll
            for (int j = 0; j < 8; ++j) {
                int   grow = rbase + j;
                float h    = acc[ni][j] + bv;
                size_t vix = (size_t)grow * ldv + gcol;
                float v    = vstate[vix];
                v = v + (h - v) * 0.5f;
                float s = (v >= 1.0f) ? 1.0f : 0.0f;
                vstate[vix] = v * (1.0f - s);
                if (sout) sout[(size_t)grow * ldsout + gcol] = (_Float16)s;
                if (fout) fout[(size_t)grow * ldfout + gcol] = s;
            }
        }
    }
}

extern "C" void kernel_launch(void* const* d_in, const int* in_sizes, int n_in,
                              void* d_out, int out_size, void* d_ws, size_t ws_size,
                              hipStream_t stream)
{
    const int B = 16384, T = 3, D = 2752, H = 256, O = 100;
    const float* dvs = (const float*)d_in[0];
    const float* W1  = (const float*)d_in[1];
    const float* b1  = (const float*)d_in[2];
    const float* W2  = (const float*)d_in[3];
    const float* b2  = (const float*)d_in[4];
    const float* W3  = (const float*)d_in[5];
    const float* b3  = (const float*)d_in[6];

    // workspace layout: v1[B,H] f32 | v2[B,H] f32 | v3[B,O] f32 | s1[B,H] f16 | s2[B,H] f16
    float*    v1 = (float*)d_ws;
    float*    v2 = v1 + (size_t)B * H;
    float*    v3 = v2 + (size_t)B * H;
    _Float16* s1 = (_Float16*)(v3 + (size_t)B * O);
    _Float16* s2 = s1 + (size_t)B * H;

    // zero membrane-potential state each launch (graph-capture safe)
    size_t zbytes = ((size_t)2 * B * H + (size_t)B * O) * sizeof(float);
    hipMemsetAsync(d_ws, 0, zbytes, stream);

    dim3 grid(B / BM), block(256);
    for (int t = 0; t < T; ++t) {
        // layer 1: x[t] (f32, strided view of dvs[B,T,D]) @ W1^T -> s1
        gemm_lif<16, true ><<<grid, block, 0, stream>>>(
            dvs + (size_t)t * D, T * D, W1, D, H, b1,
            v1, H, s1, H, nullptr, 0, D, H);
        // layer 2: s1 (f16 spikes) @ W2^T -> s2
        gemm_lif<16, false><<<grid, block, 0, stream>>>(
            s1, H, W2, H, H, b2,
            v2, H, s2, H, nullptr, 0, H, H);
        // layer 3: s2 @ W3^T -> v3 / output spikes (written only at t==2)
        gemm_lif<8, false><<<grid, block, 0, stream>>>(
            s2, H, W3, H, O, b3,
            v3, O, nullptr, 0, (t == 2) ? (float*)d_out : nullptr, O, H, O);
    }
}
